// Attention_48421461295477
// MI455X (gfx1250) — matrix operand order repaired and checked by reading, not compiled, not run on hardware
//
#include <hip/hip_runtime.h>
#include <hip/hip_bf16.h>

// MHA forward: N=2, T=4096, D=512, H=8, S=64.
// Prep:  f32->f16 convert of X; f32->f16 transposed weights (Wt[n][k]).
// GEMMs: v_wmma_f32_16x16x32_f16 everywhere (wave32).
// Flash attention: K/V tiles double-buffered in LDS via async-to-LDS
// (ASYNCcnt) when available; online softmax; f16 ctx.

typedef __attribute__((ext_vector_type(16))) _Float16 v16h;
typedef __attribute__((ext_vector_type(8)))  _Float16 v8h;
typedef __attribute__((ext_vector_type(8)))  float    v8f;
typedef __attribute__((ext_vector_type(4)))  int      v4i;

#define TT   4096
#define DD   512
#define HH   8
#define SS   64
#define NEG_INF_F (-1000000000.0f)

#define GAS __attribute__((address_space(1)))
#define LAS __attribute__((address_space(3)))

#if __has_builtin(__builtin_amdgcn_global_load_async_to_lds_b128) && \
    __has_builtin(__builtin_amdgcn_s_wait_asynccnt)
#define HAS_ASYNC 1
#endif

__device__ __forceinline__ v8f wmma_f16(v16h a, v16h b, v8f c) {
  return __builtin_amdgcn_wmma_f32_16x16x32_f16(
      /*neg_a=*/false, a, /*neg_b=*/false, b,
      /*c_mod=*/(short)0, c, /*reuse_a=*/false, /*reuse_b=*/false);
}

// ---- Fragment loaders (wave32; contiguous k order, same in A and B) ----
// A 16x32 f16: lane l -> M = l&15 ; half j -> K = 16*(l>=16) + j
__device__ __forceinline__ v16h frag_a_f16(const _Float16* A, int lda, int lane) {
  v16h a;
  const _Float16* base = A + (lane & 15) * lda + ((lane >> 4) << 4);
  #pragma unroll
  for (int p = 0; p < 8; ++p) {
    int k = p << 1;   // 0,2,4,...,14
    a[2 * p]     = base[k];
    a[2 * p + 1] = base[k + 1];
  }
  return a;
}

// B 32x16 f16 from [N][K]-layout tile: lane l -> N=l&15, halves j -> K=16*(l>=16)+j
// (32 contiguous bytes per lane).
__device__ __forceinline__ v16h frag_b_f16(const _Float16* Bt, int ldb, int lane) {
  v16h b;
  const _Float16* base = Bt + (lane & 15) * ldb + ((lane >> 4) << 4);
  #pragma unroll
  for (int j = 0; j < 16; ++j) b[j] = base[j];
  return b;
}

// B fragment gathered from a row-major [K][N] tile (no transposed copy needed).
__device__ __forceinline__ v16h frag_b_f16T(const _Float16* V, int ldv, int lane,
                                            int k0, int n0) {
  v16h b;
  const int n  = n0 + (lane & 15);
  const int kk = k0 + ((lane >> 4) << 4);
  #pragma unroll
  for (int j = 0; j < 16; ++j) b[j] = V[(kk + j) * ldv + n];
  return b;
}

// ---- Prep kernels ----
__global__ void convert_f16_kernel(const float* __restrict__ src,
                                   _Float16* __restrict__ dst) {
  const int i = (blockIdx.x * 256 + threadIdx.x) * 8;
  #pragma unroll
  for (int j = 0; j < 8; ++j) dst[i + j] = (_Float16)src[i + j];
}

// W [512][512] row-major (k,n) -> Wt[n][k] f16
__global__ void transpose_w_kernel(const float* __restrict__ W,
                                   _Float16* __restrict__ Wt) {
  const int idx = blockIdx.x * 256 + threadIdx.x;  // n*512 + k
  const int n = idx >> 9, k = idx & 511;
  Wt[idx] = (_Float16)W[k * DD + n];
}

// ---- Kernel 1: X[8192,512]f16 x Wt[512,512]f16 -> f16 out in [N,H,T,S] ----
__global__ void proj_qkv_kernel(const _Float16* __restrict__ X,
                                const _Float16* __restrict__ Wt,
                                _Float16* __restrict__ out, float scale) {
  const int lane = threadIdx.x & 31;
  const int wave = threadIdx.x >> 5;
  const int col0 = blockIdx.x * 16;              // 0..511
  const int row0 = blockIdx.y * 64 + wave * 16;  // 0..8191
  v8f acc;
  #pragma unroll
  for (int r = 0; r < 8; ++r) acc[r] = 0.0f;

  const _Float16* Abase = X  + (size_t)row0 * DD;
  const _Float16* Bbase = Wt + (size_t)col0 * DD;
  #pragma unroll 4
  for (int k0 = 0; k0 < DD; k0 += 32) {
    v16h a = frag_a_f16(Abase + k0, DD, lane);
    v16h b = frag_b_f16(Bbase + k0, DD, lane);
    acc = wmma_f16(a, b, acc);
  }
  const int c  = col0 + (lane & 15);
  const int h  = c >> 6;
  const int s  = c & 63;
  const int mo = (lane >> 4) << 3;
  #pragma unroll
  for (int r = 0; r < 8; ++r) {
    const int m = row0 + mo + r;
    const int n = m >> 12;
    const int t = m & (TT - 1);
    out[(((size_t)(n * HH + h) * TT + t) << 6) + s] = (_Float16)(acc[r] * scale);
  }
}

// Stage one 64x64 f16 tile (8KB) into LDS: 128 threads x 4 x b128.
__device__ __forceinline__ void stage_tile(const _Float16* gbase,
                                           _Float16* lbase, int tid) {
#ifdef HAS_ASYNC
  #pragma unroll
  for (int c = 0; c < 4; ++c) {
    const int off = tid * 8 + c * 1024;
    __builtin_amdgcn_global_load_async_to_lds_b128(
        (GAS v4i*)(gbase + off), (LAS v4i*)(lbase + off), 0, 0);
  }
#else
  #pragma unroll
  for (int c = 0; c < 4; ++c) {
    const int off = tid * 8 + c * 1024;
    *(v8h*)(lbase + off) = *(const v8h*)(gbase + off);
  }
#endif
}

// ---- Kernel 2: fused flash attention, one (n,h) + 64 query rows / block ----
__global__ void flash_attn_kernel(const _Float16* __restrict__ q,
                                  const _Float16* __restrict__ kmat,
                                  const _Float16* __restrict__ vmat,
                                  const float* __restrict__ mask,
                                  _Float16* __restrict__ ctx) {
  __shared__ _Float16 Kt[2][64 * SS];     // [buf][key][s] 8KB each
  __shared__ _Float16 Vt[2][64 * SS];     // [buf][key][s] 8KB each
  __shared__ _Float16 Plds[4][16 * 64];   // per-wave P tiles 8KB

  const int tid  = threadIdx.x;
  const int lane = tid & 31;
  const int wave = tid >> 5;
  const int nh   = blockIdx.y;            // n*H + h
  const int n    = nh >> 3;
  const int h    = nh & 7;
  const int qt   = blockIdx.x * 64 + wave * 16;

  const _Float16* qbase = q    + (size_t)nh * TT * SS;
  const _Float16* kbase = kmat + (size_t)nh * TT * SS;
  const _Float16* vbase = vmat + (size_t)nh * TT * SS;
  const float*    maskN = mask + (size_t)n * TT;

  const v16h aq0 = frag_a_f16(qbase + (size_t)qt * SS + 0,  SS, lane);
  const v16h aq1 = frag_a_f16(qbase + (size_t)qt * SS + 32, SS, lane);

  float mstate[8], lstate[8];
  v8f o[4];
  #pragma unroll
  for (int r = 0; r < 8; ++r) { mstate[r] = -3.0e38f; lstate[r] = 0.0f; }
  #pragma unroll
  for (int ss = 0; ss < 4; ++ss)
    #pragma unroll
    for (int r = 0; r < 8; ++r) o[ss][r] = 0.0f;

  const int mo = (lane >> 4) << 3;

  // Prologue: stage key block 0 into buffer 0.
  stage_tile(kbase, &Kt[0][0], tid);
  stage_tile(vbase, &Vt[0][0], tid);

  for (int kb = 0, it = 0; kb < TT; kb += 64, ++it) {
    const int cur = it & 1;
#ifdef HAS_ASYNC
    __builtin_amdgcn_s_wait_asynccnt(0);   // my slice of buf[cur] landed
#endif
    __syncthreads();                       // everyone's slices landed; buf[cur^1] free
    if (kb + 64 < TT) {                    // issue next block into the other buffer
      stage_tile(kbase + (size_t)(kb + 64) * SS, &Kt[cur ^ 1][0], tid);
      stage_tile(vbase + (size_t)(kb + 64) * SS, &Vt[cur ^ 1][0], tid);
    }
    const _Float16* Kl = &Kt[cur][0];
    const _Float16* Vl = &Vt[cur][0];

    // ---- S = Q K^T : B^T layout for QK^T is K's natural [key][s] layout ----
    v8f sc[4];
    #pragma unroll
    for (int sub = 0; sub < 4; ++sub) {
      const _Float16* kt = Kl + sub * 16 * SS;
      v16h b0 = frag_b_f16(kt + 0,  SS, lane);
      v16h b1 = frag_b_f16(kt + 32, SS, lane);
      v8f c;
      #pragma unroll
      for (int r = 0; r < 8; ++r) c[r] = 0.0f;
      c = wmma_f16(aq0, b0, c);
      c = wmma_f16(aq1, b1, c);
      const float bias = maskN[kb + sub * 16 + (lane & 15)] * NEG_INF_F;
      #pragma unroll
      for (int r = 0; r < 8; ++r) c[r] += bias;
      sc[sub] = c;
    }

    // ---- online softmax (rows live across 16-lane halves) ----
    float mnew[8];
    #pragma unroll
    for (int r = 0; r < 8; ++r) {
      float mx = mstate[r];
      #pragma unroll
      for (int sub = 0; sub < 4; ++sub) mx = fmaxf(mx, sc[sub][r]);
      mnew[r] = mx;
    }
    #pragma unroll
    for (int r = 0; r < 8; ++r) {
      #pragma unroll
      for (int off = 1; off < 16; off <<= 1)
        mnew[r] = fmaxf(mnew[r], __shfl_xor(mnew[r], off, 32));
    }
    float psum[8];
    #pragma unroll
    for (int r = 0; r < 8; ++r) {
      const float alpha = __expf(mstate[r] - mnew[r]);
      lstate[r] *= alpha;
      #pragma unroll
      for (int ss = 0; ss < 4; ++ss) o[ss][r] *= alpha;
      mstate[r] = mnew[r];
      psum[r] = 0.0f;
    }
    #pragma unroll
    for (int sub = 0; sub < 4; ++sub) {
      #pragma unroll
      for (int r = 0; r < 8; ++r) {
        const float p = __expf(sc[sub][r] - mnew[r]);
        psum[r] += p;
        Plds[wave][(r + mo) * 64 + sub * 16 + (lane & 15)] = (_Float16)p;
      }
    }
    #pragma unroll
    for (int r = 0; r < 8; ++r) {
      #pragma unroll
      for (int off = 1; off < 16; off <<= 1)
        psum[r] += __shfl_xor(psum[r], off, 32);
      lstate[r] += psum[r];
    }

    // ---- O += P V : A = P (wave-private LDS), B gathered from V [key][s] ----
    v16h ap0 = frag_a_f16(&Plds[wave][0],  64, lane);
    v16h ap1 = frag_a_f16(&Plds[wave][32], 64, lane);
    #pragma unroll
    for (int ss = 0; ss < 4; ++ss) {
      v16h b0 = frag_b_f16T(Vl, SS, lane, 0,  ss * 16);
      v16h b1 = frag_b_f16T(Vl, SS, lane, 32, ss * 16);
      o[ss] = wmma_f16(ap0, b0, o[ss]);
      o[ss] = wmma_f16(ap1, b1, o[ss]);
    }
  }

  // Normalize; store ctx in [N,T,H,S] (== [N*T] x [H*S] rows for the Wo GEMM)
  #pragma unroll
  for (int r = 0; r < 8; ++r) {
    const float inv = 1.0f / lstate[r];
    const int t = qt + mo + r;
    #pragma unroll
    for (int ss = 0; ss < 4; ++ss) {
      const int scol = ss * 16 + (lane & 15);
      ctx[(((size_t)n * TT + t) * HH + h) * SS + scol] = (_Float16)(o[ss][r] * inv);
    }
  }
}

// ---- Kernel 3: ctx[8192,512]f16 x WoT[512,512]f16 -> out f32 ----
__global__ void proj_out_kernel(const _Float16* __restrict__ ctx,
                                const _Float16* __restrict__ WoT,
                                float* __restrict__ out) {
  const int lane = threadIdx.x & 31;
  const int wave = threadIdx.x >> 5;
  const int col0 = blockIdx.x * 16;
  const int row0 = blockIdx.y * 64 + wave * 16;
  v8f acc;
  #pragma unroll
  for (int r = 0; r < 8; ++r) acc[r] = 0.0f;

  const _Float16* Abase = ctx + (size_t)row0 * DD;
  const _Float16* Bbase = WoT + (size_t)col0 * DD;
  #pragma unroll 4
  for (int k0 = 0; k0 < DD; k0 += 32) {
    v16h a = frag_a_f16(Abase + k0, DD, lane);
    v16h b = frag_b_f16(Bbase + k0, DD, lane);
    acc = wmma_f16(a, b, acc);
  }
  const int cidx = col0 + (lane & 15);
  const int mo   = (lane >> 4) << 3;
  #pragma unroll
  for (int r = 0; r < 8; ++r)
    out[(size_t)(row0 + mo + r) * DD + cidx] = acc[r];
}

extern "C" void kernel_launch(void* const* d_in, const int* in_sizes, int n_in,
                              void* d_out, int out_size, void* d_ws, size_t ws_size,
                              hipStream_t stream) {
  const float* query = (const float*)d_in[0];   // [2,4096,512]
  const float* refer = (const float*)d_in[1];   // [2,4096,512]
  const float* mask  = (const float*)d_in[2];   // [2,1,1,4096]
  const float* Wq    = (const float*)d_in[3];   // [512,8,64]
  const float* Wk    = (const float*)d_in[4];
  const float* Wv    = (const float*)d_in[5];
  const float* Wo    = (const float*)d_in[6];   // [8,64,512]
  float* out = (float*)d_out;

  const size_t E = (size_t)2 * HH * TT * SS;    // 4M halves per tensor
  const size_t WE = (size_t)DD * DD;            // 256K halves per weight
  _Float16* p   = (_Float16*)d_ws;
  _Float16* qb  = p;  p += E;
  _Float16* kb  = p;  p += E;
  _Float16* vb  = p;  p += E;
  _Float16* cb  = p;  p += E;
  _Float16* xq  = p;  p += E;
  _Float16* xr  = p;  p += E;
  _Float16* wqt = p;  p += WE;
  _Float16* wkt = p;  p += WE;
  _Float16* wvt = p;  p += WE;
  _Float16* wot = p;  p += WE;                  // ~50 MB total

  // Prep: f16 inputs + transposed f16 weights (one-time conversion cost).
  convert_f16_kernel<<<dim3(E / 2048), 256, 0, stream>>>(query, xq);
  convert_f16_kernel<<<dim3(E / 2048), 256, 0, stream>>>(refer, xr);
  transpose_w_kernel<<<dim3(WE / 256), 256, 0, stream>>>(Wq, wqt);
  transpose_w_kernel<<<dim3(WE / 256), 256, 0, stream>>>(Wk, wkt);
  transpose_w_kernel<<<dim3(WE / 256), 256, 0, stream>>>(Wv, wvt);
  transpose_w_kernel<<<dim3(WE / 256), 256, 0, stream>>>(Wo, wot);

  dim3 pgrid(DD / 16, (2 * TT) / 64);  // (32,128)
  dim3 pblk(128);
  proj_qkv_kernel<<<pgrid, pblk, 0, stream>>>(xq, wqt, qb, 0.125f); // S^-0.5
  proj_qkv_kernel<<<pgrid, pblk, 0, stream>>>(xr, wkt, kb, 1.0f);
  proj_qkv_kernel<<<pgrid, pblk, 0, stream>>>(xr, wvt, vb, 1.0f);

  flash_attn_kernel<<<dim3(TT / 64, 2 * HH), 128, 0, stream>>>(qb, kb, vb, mask, cb);

  proj_out_kernel<<<pgrid, pblk, 0, stream>>>(cb, wot, out);
}
